// MultiHeadAttention_19000935317593
// MI455X (gfx1250) — matrix operand order repaired and checked by reading, not compiled, not run on hardware
//
#include <hip/hip_runtime.h>
#include <stdint.h>
#include <stddef.h>

typedef __attribute__((ext_vector_type(16))) __bf16 v16bf;
typedef __attribute__((ext_vector_type(8)))  __bf16 v8bf;
typedef __attribute__((ext_vector_type(4)))  __bf16 v4bf;
typedef __attribute__((ext_vector_type(8)))  float  v8f;

// Matches the pointee type of the async-to-LDS builtin's pointer params.
typedef int i4v __attribute__((vector_size(16)));

union Frag {
    v16bf v;
    v8bf  h[2];
    __bf16 e[16];
};

__device__ __forceinline__ v16bf zero_frag() {
    Frag f;
#pragma unroll
    for (int i = 0; i < 16; ++i) f.e[i] = (__bf16)0.0f;
    return f.v;
}

// --------------------------- CDNA5 async global->LDS ----------------------
#if defined(__gfx1250__) && __has_builtin(__builtin_amdgcn_global_load_async_to_lds_b128)
#define HAVE_ASYNC_LDS 1
#else
#define HAVE_ASYNC_LDS 0
#endif

// Copy 16 bytes (8 bf16) global -> LDS, asynchronously if available.
__device__ __forceinline__ void async_copy16(const __bf16* g, __bf16* l) {
#if HAVE_ASYNC_LDS
    __builtin_amdgcn_global_load_async_to_lds_b128(
        (__attribute__((address_space(1))) i4v*)g,
        (__attribute__((address_space(3))) i4v*)l, 0, 0);
#else
    *(v8bf*)l = *(const v8bf*)g;
#endif
}

template <int N>
__device__ __forceinline__ void wait_async() {
#if HAVE_ASYNC_LDS
#if __has_builtin(__builtin_amdgcn_s_wait_asynccnt)
    __builtin_amdgcn_s_wait_asynccnt(N);
#else
    asm volatile("s_wait_asynccnt %0" ::"i"(N) : "memory");
#endif
#endif
}

// ---------------------------------------------------------------------------
// Kernel 1: pack W [1024x1024] f32 into bf16 B-fragment-major layout.
// Wp[((C*64 + J)*32 + lane)*16 + q] = W[(J*16 + lane%16)*1024 + C*32 + kk]
//   where kk = (q&7) + (lane>=16 ? 8 : 0) + (q>=8 ? 16 : 0).
// C = K-chunk (0..31), J = 16-col tile (0..63). Each GEMM B-fragment is then
// one fully coalesced 32B load per lane (1KB contiguous per wave).
// ---------------------------------------------------------------------------
__global__ __launch_bounds__(256) void pack_w_kernel(const float* __restrict__ W,
                                                     __bf16* __restrict__ Wp) {
    const int t = blockIdx.x * 256 + threadIdx.x;   // 0 .. 262143
#pragma unroll
    for (int i = 0; i < 4; ++i) {
        const int idx  = t * 4 + i;                 // 0 .. 1048575
        const int q    = idx & 15;
        const int lane = (idx >> 4) & 31;
        const int J    = (idx >> 9) & 63;
        const int C    = idx >> 15;
        const int row  = J * 16 + (lane & 15);
        const int kk   = (q & 7) + ((lane < 16) ? 0 : 8) + ((q & 8) ? 16 : 0);
        Wp[idx] = (__bf16)W[(size_t)row * 1024 + C * 32 + kk];
    }
}

// ---------------------------------------------------------------------------
// Kernel 2: per-token head-to-head attention, one wave (32 lanes) per token.
//   S = q k^T / 8   -> 2x v_wmma_f32_16x16x32_bf16 (K=64)
//   P = softmax(S)  -> lane-shuffle reductions (rows live across 16 lanes)
//   O = P v         -> 4x wmma (N=64 in 4 tiles, K=16 zero-padded to 32)
// O written as bf16 into scrambled layout: row = b*4096 + h*256 + n/16,
// col = (n%16)*64 + dd, ready for the projection GEMM.
// ---------------------------------------------------------------------------
__global__ __launch_bounds__(32) void attn_kernel(const float* __restrict__ x,
                                                  const float* __restrict__ y,
                                                  __bf16* __restrict__ Abf) {
    __shared__ __align__(32) __bf16 qb[1024];   // q row [16 heads][64]
    __shared__ __align__(32) __bf16 kb[1024];   // k row [16 heads][64]
    __shared__ __align__(32) __bf16 vt[1024];   // v transposed [64 dd][16 g]
    __shared__ __align__(32) __bf16 ps[256];    // softmax(S) [16 h][16 g]

    const int tok  = blockIdx.x;           // 0..16383
    const int b    = tok >> 12;
    const int n    = tok & 4095;
    const int lane = threadIdx.x;          // 0..31
    const int lrow = lane & 15;
    const bool hi  = lane >= 16;

    // Stage x,y rows (1024 f32 each) -> bf16 LDS. 8 float4 per lane per row.
    const float4* xr = (const float4*)(x + (size_t)tok * 1024);
    const float4* yr = (const float4*)(y + (size_t)tok * 1024);
#pragma unroll
    for (int i = 0; i < 8; ++i) {
        float4 fx = xr[lane * 8 + i];
        float4 fy = yr[lane * 8 + i];
        v4bf bx = {(__bf16)fx.x, (__bf16)fx.y, (__bf16)fx.z, (__bf16)fx.w};
        v4bf by = {(__bf16)fy.x, (__bf16)fy.y, (__bf16)fy.z, (__bf16)fy.w};
        *(v4bf*)(qb + lane * 32 + i * 4) = bx;
        *(v4bf*)(kb + lane * 32 + i * 4) = by;
    }
    __syncthreads();

    // Build vt[dd][g] = v[g][dd] (v == k row). 32 elems per lane.
#pragma unroll
    for (int i = 0; i < 32; ++i) {
        const int e = lane * 32 + i;
        vt[e] = kb[(e & 15) * 64 + (e >> 4)];
    }

    // S = q k^T  (M=16 heads of q, N=16 heads of k, K=64 in two chunks of 32)
    v8f s = {};
#pragma unroll
    for (int c = 0; c < 2; ++c) {
        Frag a, bk;
        const int ab = lrow * 64 + c * 32 + (hi ? 8 : 0);
        a.h[0] = *(const v8bf*)(qb + ab);        // K {0..7} / {8..15}
        a.h[1] = *(const v8bf*)(qb + ab + 16);   // K {16..23} / {24..31}
        const int bb = lrow * 64 + c * 32 + (hi ? 8 : 0);
        bk.h[0] = *(const v8bf*)(kb + bb);       // K {0..7} / {8..15}
        bk.h[1] = *(const v8bf*)(kb + bb + 16);  // K {16..23} / {24..31}
        s = __builtin_amdgcn_wmma_f32_16x16x32_bf16(false, a.v, false, bk.v,
                                                    (short)0, s, false, false);
    }

    // Softmax over g (the N/lane axis). Row (j, half) spans 16 lanes.
    float p[8];
#pragma unroll
    for (int j = 0; j < 8; ++j) {
        const float sv = s[j] * 0.125f;          // 1/sqrt(64)
        float mx = sv;
#pragma unroll
        for (int m = 8; m >= 1; m >>= 1) mx = fmaxf(mx, __shfl_xor(mx, m, 32));
        const float ev = __expf(sv - mx);
        float sum = ev;
#pragma unroll
        for (int m = 8; m >= 1; m >>= 1) sum += __shfl_xor(sum, m, 32);
        p[j] = ev / sum;
    }

    // P -> LDS (bf16) so it can be regathered in A-fragment layout.
#pragma unroll
    for (int j = 0; j < 8; ++j) {
        const int m = hi ? j + 8 : j;
        ps[m * 16 + lrow] = (__bf16)p[j];
    }
    __syncthreads();

    // A fragment of P: K = g (only 0..15 real; 16..31 zero-padded).
    Frag ap;
    ap.h[0] = *(const v8bf*)(ps + lrow * 16 + (hi ? 8 : 0));
#pragma unroll
    for (int i = 8; i < 16; ++i) ap.e[i] = (__bf16)0.0f;

    const int rbase  = b * 4096 + (n >> 4);
    const int cbase0 = (n & 15) * 64;

#pragma unroll
    for (int t = 0; t < 4; ++t) {
        Frag bv;
        bv.h[0] = *(const v8bf*)(vt + (16 * t + lrow) * 16 + (hi ? 8 : 0)); // K 0..7 / 8..15
#pragma unroll
        for (int i = 8; i < 16; ++i) bv.e[i] = (__bf16)0.0f;               // K 16..31 pad
        v8f o = {};
        o = __builtin_amdgcn_wmma_f32_16x16x32_bf16(false, ap.v, false, bv.v,
                                                    (short)0, o, false, false);
        const int col = cbase0 + 16 * t + lrow;
#pragma unroll
        for (int j = 0; j < 8; ++j) {
            const int m = hi ? j + 8 : j;                 // head index
            Abf[(size_t)(rbase + m * 256) * 1024 + col] = (__bf16)o[j];
        }
    }
}

// ---------------------------------------------------------------------------
// Kernel 3: projection GEMM. out[16384,1024] = A'(bf16) * W^T + bias, f32 out.
// Block = 256 threads = 8 waves. Tile: 128 rows x 64 cols. K = 1024 in 32
// chunks of 32. A tile double-buffered in LDS via async global->LDS copies
// (ASYNCcnt pipeline: prefetch chunk k+1 while computing chunk k). B fragments
// read directly from L2-resident packed W (one 32B load per lane per tile).
// 4x wmma_f32_16x16x32_bf16 per wave per K-chunk.
// ---------------------------------------------------------------------------
__global__ __launch_bounds__(256) void proj_kernel(const __bf16* __restrict__ Ap,
                                                   const __bf16* __restrict__ Wp,
                                                   const float* __restrict__ bias,
                                                   float* __restrict__ out) {
    __shared__ __align__(32) __bf16 As[2][128 * 32];

    const int tid  = threadIdx.x;
    const int wave = tid >> 5;             // 0..7 -> m-tile within block
    const int lane = tid & 31;
    const int lrow = lane & 15;
    const bool hi  = lane >= 16;
    const int r0 = blockIdx.x * 128;
    const int n0 = blockIdx.y * 64;

    v8f acc[4];
#pragma unroll
    for (int t = 0; t < 4; ++t) { v8f z = {}; acc[t] = z; }

    // Each thread stages 32B per chunk as two 16B async copies.
    const int srow = tid >> 1;             // 0..127
    const int sseg = (tid & 1) * 16;       // 0 or 16 elements
    const __bf16* gbase = Ap + (size_t)(r0 + srow) * 1024 + sseg;

    // Prologue: kick off chunk 0 into buffer 0.
    async_copy16(gbase, &As[0][srow * 32 + sseg]);
    async_copy16(gbase + 8, &As[0][srow * 32 + sseg + 8]);

    for (int kc = 0; kc < 32; ++kc) {
        const int cur = kc & 1;

        // Prefetch next chunk into the other buffer (WAR on that buffer is
        // protected by the barrier at the end of the previous iteration).
        if (kc + 1 < 32) {
            const __bf16* g = gbase + (kc + 1) * 32;
            async_copy16(g, &As[cur ^ 1][srow * 32 + sseg]);
            async_copy16(g + 8, &As[cur ^ 1][srow * 32 + sseg + 8]);
            wait_async<2>();   // own loads for 'cur' retired; prefetch in flight
        } else {
            wait_async<0>();
        }
        __syncthreads();       // whole As[cur] tile visible to all 8 waves

        Frag a;
        const int ab = (wave * 16 + lrow) * 32 + (hi ? 8 : 0);
        a.h[0] = *(const v8bf*)(&As[cur][ab]);
        a.h[1] = *(const v8bf*)(&As[cur][ab + 16]);

        // Packed-W fragment base: ((kc*64 + J)*32 + lane)*16, J = n0/16 + t.
        const size_t wb = ((size_t)kc * 64 + (n0 >> 4)) * 512 + (size_t)lane * 16;
        const v16bf b0 = *(const v16bf*)(Wp + wb);
        const v16bf b1 = *(const v16bf*)(Wp + wb + 512);
        const v16bf b2 = *(const v16bf*)(Wp + wb + 1024);
        const v16bf b3 = *(const v16bf*)(Wp + wb + 1536);

        acc[0] = __builtin_amdgcn_wmma_f32_16x16x32_bf16(false, a.v, false, b0,
                                                         (short)0, acc[0], false, false);
        acc[1] = __builtin_amdgcn_wmma_f32_16x16x32_bf16(false, a.v, false, b1,
                                                         (short)0, acc[1], false, false);
        acc[2] = __builtin_amdgcn_wmma_f32_16x16x32_bf16(false, a.v, false, b2,
                                                         (short)0, acc[2], false, false);
        acc[3] = __builtin_amdgcn_wmma_f32_16x16x32_bf16(false, a.v, false, b3,
                                                         (short)0, acc[3], false, false);

        __syncthreads();       // all waves done reading As[cur] before it is
                               // overwritten by the prefetch of chunk kc+2
    }

    // Epilogue: add bias, store f32 (lanes 0..15 contiguous 64B per store).
#pragma unroll
    for (int t = 0; t < 4; ++t) {
        const int col = n0 + 16 * t + lrow;
        const float bv = bias[col];
#pragma unroll
        for (int j = 0; j < 8; ++j) {
            const int row = r0 + wave * 16 + (hi ? j + 8 : j);
            out[(size_t)row * 1024 + col] = acc[t][j] + bv;
        }
    }
}

// ---------------------------------------------------------------------------
extern "C" void kernel_launch(void* const* d_in, const int* in_sizes, int n_in,
                              void* d_out, int out_size, void* d_ws, size_t ws_size,
                              hipStream_t stream) {
    (void)in_sizes; (void)n_in; (void)out_size; (void)ws_size;

    const float* x    = (const float*)d_in[0];   // [4,4096,1024]
    const float* y    = (const float*)d_in[1];   // [4,4096,1024]
    const float* W    = (const float*)d_in[2];   // [1024,1024]
    const float* bias = (const float*)d_in[3];   // [1024]
    float* out = (float*)d_out;                  // [4,4096,1024]

    __bf16* Abf = (__bf16*)d_ws;                                     // 32 MB
    __bf16* Wp  = (__bf16*)((char*)d_ws + (size_t)16384 * 1024 * 2); // 2 MB

    pack_w_kernel<<<1024, 256, 0, stream>>>(W, Wp);
    attn_kernel<<<16384, 32, 0, stream>>>(x, y, Abf);
    proj_kernel<<<dim3(128, 16), 256, 0, stream>>>(Abf, Wp, bias, out);
}
